// MyMSA_22239340658969
// MI455X (gfx1250) — matrix operand order repaired and checked by reading, not compiled, not run on hardware
//
#include <hip/hip_runtime.h>

#define DD  1024
#define HH  16
#define DHH 64
#define SS  2048

typedef __attribute__((ext_vector_type(16))) _Float16 v16h;
typedef __attribute__((ext_vector_type(8)))  _Float16 v8h;
typedef __attribute__((ext_vector_type(8)))  float    v8f;
typedef int v4i_gcc __attribute__((vector_size(16)));   // matches builtin's V4i

// ---- gfx1250 async global->LDS path (guarded so we always compile) ----------
#if defined(__has_builtin)
#  if __has_builtin(__builtin_amdgcn_global_load_async_to_lds_b128)
#    define USE_ASYNC_LDS 1
#  endif
#  if __has_builtin(__builtin_amdgcn_s_wait_asynccnt)
#    define HAVE_WAIT_ASYNC_BUILTIN 1
#  endif
#endif
#ifndef USE_ASYNC_LDS
#  define USE_ASYNC_LDS 0
#endif
#ifndef HAVE_WAIT_ASYNC_BUILTIN
#  define HAVE_WAIT_ASYNC_BUILTIN 0
#endif

__device__ __forceinline__ void async_cp16(_Float16* l, const _Float16* g) {
#if USE_ASYNC_LDS
  __builtin_amdgcn_global_load_async_to_lds_b128(
      (__attribute__((address_space(1))) v4i_gcc*)(g),
      (__attribute__((address_space(3))) v4i_gcc*)(l), 0, 0);
#else
  *(v8h*)l = *(const v8h*)g;   // synchronous fallback: load + ds_store
#endif
}

__device__ __forceinline__ void wait_async_lds() {
#if USE_ASYNC_LDS
#if HAVE_WAIT_ASYNC_BUILTIN
  __builtin_amdgcn_s_wait_asynccnt(0);
#else
  asm volatile("s_wait_asynccnt 0x0" ::: "memory");
#endif
#endif
}

__device__ __forceinline__ v8f wmma16(v16h a, v16h b, v8f c) {
  // D(f32 16x16) = A(f16 16x32) * B(f16 32x16) + C
  return __builtin_amdgcn_wmma_f32_16x16x32_f16(false, a, false, b, (short)0, c,
                                                false, false);
}

// ---- Fragment loaders (wave32, contiguous k = 16*(lane>>4) + s order, same in A and B) -----------------
// A fragment 16x32 f16: lane m=lane&15, khalf=lane>>4
//   elems 0..15 = row[ 16*khalf .. +15 ]
__device__ __forceinline__ v16h load_a_h(const _Float16* base, int ld, int lane) {
  int m = lane & 15, h8 = (lane >> 4) * 16;
  const _Float16* r = base + (size_t)m * ld;
  v8h lo = *(const v8h*)(r + h8);
  v8h hi = *(const v8h*)(r + 8 + h8);
  v16h a;
#pragma unroll
  for (int i = 0; i < 8; ++i) { a[i] = lo[i]; a[8 + i] = hi[i]; }
  return a;
}

__device__ __forceinline__ v16h load_a_f(const float* base, int ld, int lane) {
  int m = lane & 15, h8 = (lane >> 4) * 16;
  const float* r = base + (size_t)m * ld;
  v16h a;
#pragma unroll
  for (int i = 0; i < 8; ++i) {
    a[i]     = (_Float16)r[h8 + i];
    a[8 + i] = (_Float16)r[8 + h8 + i];
  }
  return a;
}

// B fragment 32x16: lane n=lane&15 reads 16 contiguous f16 of "B-transposed"
// row n at column offset 16*(lane>>4).
__device__ __forceinline__ v16h load_b_h(const _Float16* base, int ld, int lane) {
  int n = lane & 15, k16 = (lane >> 4) * 16;
  const _Float16* r = base + (size_t)n * ld + k16;
  return *(const v16h*)r;
}

__device__ __forceinline__ v16h load_b_f(const float* base, int ld, int lane) {
  int n = lane & 15, k16 = (lane >> 4) * 16;
  const float* r = base + (size_t)n * ld + k16;
  v16h b;
#pragma unroll
  for (int i = 0; i < 16; ++i) b[i] = (_Float16)r[i];
  return b;
}

// ---- Kernel 1: per-head QKV projection, fp32 -> f16 -------------------------
// Q (pre-scaled by 1/sqrt(DH)), K row-major (nh,S,DH); V transposed (nh,DH,S).
__global__ __launch_bounds__(256) void qkv_proj_kernel(
    const float* __restrict__ x,
    const float* __restrict__ Wq, const float* __restrict__ Wk,
    const float* __restrict__ Wv,
    const float* __restrict__ bq, const float* __restrict__ bk,
    const float* __restrict__ bv,
    _Float16* __restrict__ qws, _Float16* __restrict__ kws,
    _Float16* __restrict__ vtws) {
  const int TS = SS / 128;
  int nh = blockIdx.x / TS, st = blockIdx.x % TS;
  int n = nh / HH, h = nh % HH;
  int wave = threadIdx.x >> 5, lane = threadIdx.x & 31;
  int s0 = st * 128 + wave * 16;

  const float* xb = x + ((size_t)n * SS + s0) * DD + h * DHH;
  v16h a0 = load_a_f(xb,      DD, lane);
  v16h a1 = load_a_f(xb + 32, DD, lane);

  int nc = lane & 15, half = lane >> 4;

  const float* Ws[3] = {Wq, Wk, Wv};
  const float* bs[3] = {bq, bk, bv};
#pragma unroll
  for (int p = 0; p < 3; ++p) {
    const float* Wp = Ws[p] + (size_t)h * DHH * DHH;
    const float* bp = bs[p] + h * DHH;
#pragma unroll
    for (int t = 0; t < 4; ++t) {
      int e0 = t * 16;
      v16h b0 = load_b_f(Wp + (size_t)e0 * DHH,      DHH, lane);
      v16h b1 = load_b_f(Wp + (size_t)e0 * DHH + 32, DHH, lane);
      v8f c = {};
      c = wmma16(a0, b0, c);
      c = wmma16(a1, b1, c);
      float bb = bp[e0 + nc];
      if (p == 0) {                        // Q: fold in softmax scale 1/8
        _Float16* o = qws + ((size_t)nh * SS + s0) * DHH + e0 + nc;
#pragma unroll
        for (int j = 0; j < 8; ++j)
          o[(size_t)(j + 8 * half) * DHH] = (_Float16)((c[j] + bb) * 0.125f);
      } else if (p == 1) {                 // K row-major
        _Float16* o = kws + ((size_t)nh * SS + s0) * DHH + e0 + nc;
#pragma unroll
        for (int j = 0; j < 8; ++j)
          o[(size_t)(j + 8 * half) * DHH] = (_Float16)(c[j] + bb);
      } else {                             // V transposed (DH, S)
        _Float16* vt = vtws + ((size_t)nh * DHH + e0 + nc) * SS + s0;
#pragma unroll
        for (int j = 0; j < 8; ++j)
          vt[j + 8 * half] = (_Float16)(c[j] + bb);
      }
    }
  }
}

// ---- Kernel 2: flash attention, 64-key blocks, async-LDS double buffering ---
__global__ __launch_bounds__(256) void flash_attn_kernel(
    const _Float16* __restrict__ qws, const _Float16* __restrict__ kws,
    const _Float16* __restrict__ vtws, float* __restrict__ out) {
  const int TS = SS / 128;
  int nh = blockIdx.x / TS, st = blockIdx.x % TS;
  int n = nh / HH, h = nh % HH;
  int wave = threadIdx.x >> 5, lane = threadIdx.x & 31;
  int tid = threadIdx.x;
  int q0 = st * 128 + wave * 16;

  const _Float16* qb = qws  + ((size_t)nh * SS + q0) * DHH;
  const _Float16* kb = kws  + (size_t)nh * SS * DHH;
  const _Float16* vb = vtws + (size_t)nh * DHH * SS;

  // Double-buffered K/V tiles shared by all 8 waves + per-wave P scratch.
  __shared__ _Float16 kbuf[2][64 * DHH];     // 64 keys x 64 d   (2 x 8 KB)
  __shared__ _Float16 vbuf[2][DHH * 64];     // 64 d   x 64 keys (2 x 8 KB)
  __shared__ _Float16 plds[8][16 * 64];      // per-wave P tile  (16 KB)
  _Float16* pl = plds[wave];

  // Stage one 64-key block: K rows are one contiguous 8KB run; V^T rows are
  // 128B runs at stride S. 512 16-byte chunks each; 2 per thread per tile.
  auto stage = [&](int buf, int k0) {
    const _Float16* gk = kb + (size_t)k0 * DHH;
#pragma unroll
    for (int r = 0; r < 2; ++r) {
      int c = tid + r * 256;
      async_cp16(kbuf[buf] + c * 8, gk + c * 8);
      int row = c >> 3, cc = c & 7;
      async_cp16(vbuf[buf] + row * 64 + cc * 8,
                 vb + (size_t)row * SS + k0 + cc * 8);
    }
  };

  v16h aq0 = load_a_h(qb,      DHH, lane);
  v16h aq1 = load_a_h(qb + 32, DHH, lane);

  v8f o[4] = {{}, {}, {}, {}};
  float mi[8], li[8];
#pragma unroll
  for (int j = 0; j < 8; ++j) { mi[j] = -3.0e38f; li[j] = 0.0f; }

  int nc = lane & 15, half = lane >> 4;

  stage(0, 0);
  for (int it = 0, k0 = 0; k0 < SS; k0 += 64, ++it) {
    int buf = it & 1;
    wait_async_lds();       // own async copies for `buf` have landed
    __syncthreads();        // everyone's copies landed; prev reads of buf^1 done
    if (k0 + 64 < SS) stage(buf ^ 1, k0 + 64);

    const _Float16* kt = kbuf[buf];
    const _Float16* vt = vbuf[buf];

    // ---- scores: 4 key tiles x 2 k-steps (Q pre-scaled by 1/8) ----
    v8f s[4];
#pragma unroll
    for (int t = 0; t < 4; ++t) {
      v8f c = {};
      c = wmma16(aq0, load_b_h(kt + t * 16 * DHH,      DHH, lane), c);
      c = wmma16(aq1, load_b_h(kt + t * 16 * DHH + 32, DHH, lane), c);
      s[t] = c;
    }

    // ---- online softmax over 64 keys ----
#pragma unroll
    for (int j = 0; j < 8; ++j) {
      float sa = s[0][j], sb = s[1][j], sc = s[2][j], sd = s[3][j];
      float mx = fmaxf(fmaxf(sa, sb), fmaxf(sc, sd));
      mx = fmaxf(mx, __shfl_xor(mx, 1, 32));
      mx = fmaxf(mx, __shfl_xor(mx, 2, 32));
      mx = fmaxf(mx, __shfl_xor(mx, 4, 32));
      mx = fmaxf(mx, __shfl_xor(mx, 8, 32));
      float mnew  = fmaxf(mi[j], mx);
      float alpha = __expf(mi[j] - mnew);
      float p0 = __expf(sa - mnew);
      float p1 = __expf(sb - mnew);
      float p2 = __expf(sc - mnew);
      float p3 = __expf(sd - mnew);
      float rs = (p0 + p1) + (p2 + p3);
      rs += __shfl_xor(rs, 1, 32);
      rs += __shfl_xor(rs, 2, 32);
      rs += __shfl_xor(rs, 4, 32);
      rs += __shfl_xor(rs, 8, 32);
      li[j] = li[j] * alpha + rs;
      mi[j] = mnew;
      o[0][j] *= alpha; o[1][j] *= alpha; o[2][j] *= alpha; o[3][j] *= alpha;
      int row = (j + 8 * half) * 64 + nc;
      pl[row]      = (_Float16)p0;
      pl[row + 16] = (_Float16)p1;
      pl[row + 32] = (_Float16)p2;
      pl[row + 48] = (_Float16)p3;
    }

    // ---- P(16x64) @ V(64x64): intra-wave LDS transpose round-trip ----
    v16h pa0 = load_a_h(pl,      64, lane);    // keys  0..31
    v16h pa1 = load_a_h(pl + 32, 64, lane);    // keys 32..63
#pragma unroll
    for (int t = 0; t < 4; ++t) {
      o[t] = wmma16(pa0, load_b_h(vt + t * 16 * 64,      64, lane), o[t]);
      o[t] = wmma16(pa1, load_b_h(vt + t * 16 * 64 + 32, 64, lane), o[t]);
    }
  }

  // ---- epilogue: normalize, scatter into (N, S, H*DH) fp32 ----
  float* ob = out + ((size_t)n * SS + q0) * DD + h * DHH + nc;
#pragma unroll
  for (int j = 0; j < 8; ++j) {
    float inv = 1.0f / li[j];
    size_t row = (size_t)(j + 8 * half) * DD;
    ob[row +  0] = o[0][j] * inv;
    ob[row + 16] = o[1][j] * inv;
    ob[row + 32] = o[2][j] * inv;
    ob[row + 48] = o[3][j] * inv;
  }
}

extern "C" void kernel_launch(void* const* d_in, const int* in_sizes, int n_in,
                              void* d_out, int out_size, void* d_ws, size_t ws_size,
                              hipStream_t stream) {
  const float* seq = (const float*)d_in[0];
  const float* Wq  = (const float*)d_in[1];
  const float* Wk  = (const float*)d_in[2];
  const float* Wv  = (const float*)d_in[3];
  const float* bq  = (const float*)d_in[4];
  const float* bk  = (const float*)d_in[5];
  const float* bv  = (const float*)d_in[6];

  int N = in_sizes[0] / (SS * DD);         // 4

  size_t per = (size_t)N * HH * SS * DHH;  // 8,388,608 elements
  _Float16* qws  = (_Float16*)d_ws;        // 48 MB total workspace
  _Float16* kws  = qws + per;
  _Float16* vtws = kws + per;

  int blocks = N * HH * (SS / 128);        // 1024 blocks, 8 waves each
  qkv_proj_kernel<<<blocks, 256, 0, stream>>>(seq, Wq, Wk, Wv, bq, bk, bv,
                                              qws, kws, vtws);
  flash_attn_kernel<<<blocks, 256, 0, stream>>>(qws, kws, vtws, (float*)d_out);
}
